// GatedGN_65833258713647
// MI455X (gfx1250) — compile-verified, matrix operand order checked by reading, not run on hardware
//
#include <hip/hip_runtime.h>
#include <hip/hip_bf16.h>

typedef _Float16 f16;
typedef __attribute__((ext_vector_type(16))) _Float16 v16h;
typedef __attribute__((ext_vector_type(8)))  _Float16 v8h;
typedef __attribute__((ext_vector_type(8)))  float    v8f;

namespace {
constexpr int KN = 10;      // nodes per graph
constexpr int DI = 14;      // input features
constexpr int NE = 100;     // edges per graph

// ---- LDS layout (bytes), all 16B aligned ----
// sX row layout: [towers(14) | pad(2) | h(64)], stride 80 halfs, 16 rows
constexpr int OFF_X    = 0;                  // 16*80 f16 = 2560
constexpr int OFF_G16  = 2560;               // 64 f16
constexpr int OFF_E    = 2688;               // 112*64 f16 = 14336
constexpr int OFF_H    = 17024;              // 16*64 f32 = 4096
constexpr int OFF_HS   = 21120;              // 64 f32
constexpr int OFF_ES   = 21376;              // 64 f32
constexpr int OFF_GV   = 21632;              // 64 f32
constexpr int OFF_SCR  = 21888;              // union scratch
// edge phase
constexpr int OFF_RE   = OFF_SCR;            // 112*64 f16 (resets*e / mlp tmp2)
constexpr int OFF_UPD  = OFF_SCR + 14336;    // 112*64 f16 (update gate)
constexpr int OFF_T1   = OFF_SCR + 28672;    // 112*64 f16 (mlp tmp1)
// node phase (aliases edge scratch)
constexpr int OFF_NS   = OFF_SCR;            // 10*64 f32
constexpr int OFF_EDG  = OFF_SCR + 2560;     // 16*64 f16
constexpr int OFF_RH   = OFF_SCR + 4608;     // 16*64 f16
constexpr int OFF_UN   = OFF_SCR + 6656;     // 16*64 f16
constexpr int OFF_NT1  = OFF_SCR + 8704;     // 16*64 f16
constexpr int OFF_NT2  = OFF_SCR + 10752;    // 16*64 f16
// global phase (aliases scratch after node phase)
constexpr int OFF_UG   = OFF_SCR;            // 64 f32
constexpr int OFF_TG   = OFF_SCR + 256;      // 64 f32
constexpr int OFF_T2G  = OFF_SCR + 512;      // 64 f32
constexpr int SMEM_BYTES = OFF_SCR + 43008;  // 64896 B

// ---- packed weight fragments in d_ws (units: f16 halfs) ----
// one fragment = 32 lanes * 16 halfs = 512 halfs
constexpr int WMRU = 0;                    // [M_reset;M_update] K=288 N=128 : 9*8 frags
constexpr int WM0  = WMRU + 9*8*512;       // M[0]  K=288 N=64 : 9*4
constexpr int WM1  = WM0  + 9*4*512;       // M[1]  K=64  N=64 : 2*4
constexpr int WM2  = WM1  + 2*4*512;
constexpr int WURU = WM2  + 2*4*512;       // [U_reset;U_update] K=224 N=128 : 7*8
constexpr int WU0  = WURU + 7*8*512;       // U[0]  K=224 N=64 : 7*4
constexpr int WU1  = WU0  + 7*4*512;
constexpr int WU2  = WU1  + 2*4*512;
} // namespace

struct Params { const void* p[48]; };

__device__ __forceinline__ float sigm(float x){ return 1.f/(1.f + __expf(-x)); }

__device__ __forceinline__ v16h cat8(v8h lo, v8h hi){
  union { v16h v; v8h h[2]; } u;
  u.h[0] = lo; u.h[1] = hi;
  return u.v;
}

__device__ __forceinline__ v16h load_bfrag(const f16* __restrict__ base, int frag, int lane){
  return *(const v16h*)(base + (size_t)frag*512 + (size_t)lane*16);
}

// branch-free chunk pointer into virtual edge row:
// [x_i(80) | x_j(80) | g(64) | e-part(64)]  (K0 multiple of 8, chunks never cross segments)
__device__ __forceinline__ const f16* edge_ptr(const f16* sX, const f16* sG16, const f16* sEp,
                                               int i, int j, int r, int K0){
  const f16* p;
  if (K0 < 80)       p = sX + i*80 + K0;
  else if (K0 < 160) p = sX + j*80 + (K0 - 80);
  else if (K0 < 224) p = sG16 + (K0 - 160);
  else               p = sEp + r*64 + (K0 - 224);
  return p;
}

__device__ __forceinline__ v16h gather_edge_A(const f16* sX, const f16* sG16, const f16* sEp,
                                              int i, int j, int r, int kt, int lane){
  const int K0 = kt*32 + ((lane>>4)&1)*8;
  v8h lo = *(const v8h*)edge_ptr(sX, sG16, sEp, i, j, r, K0);
  v8h hi = *(const v8h*)edge_ptr(sX, sG16, sEp, i, j, r, K0 + 16);
  return cat8(lo, hi);
}

// virtual node row: [towers+pad(16) | edges(64) | g(64) | h-part(64)] (pad K to 224)
__device__ __forceinline__ const f16* node_ptr(const f16* sX, const f16* sEdg, const f16* sG16,
                                               const f16* hsrc, int hstride, int r, int K0){
  const f16* p;
  if (K0 < 16)       p = sX + r*80 + K0;
  else if (K0 < 80)  p = sEdg + r*64 + (K0 - 16);
  else if (K0 < 144) p = sG16 + (K0 - 80);
  else { int o = K0 - 144; if (o > 56) o = 56; p = hsrc + r*hstride + o; }
  return p;
}

__device__ __forceinline__ v16h gather_node_A(const f16* sX, const f16* sEdg, const f16* sG16,
                                              const f16* hsrc, int hstride, int r, int kt, int lane){
  const int K0 = kt*32 + ((lane>>4)&1)*8;
  v8h lo = *(const v8h*)node_ptr(sX, sEdg, sG16, hsrc, hstride, r, K0);
  v8h hi = *(const v8h*)node_ptr(sX, sEdg, sG16, hsrc, hstride, r, K0 + 16);
  return cat8(lo, hi);
}

// plain rows x 64 f16 matrix, K=64
__device__ __forceinline__ v16h gather_row64_A(const f16* M, int r, int kt, int lane){
  const f16* p = M + r*64 + kt*32 + ((lane>>4)&1)*8;
  return cat8(*(const v8h*)p, *(const v8h*)(p + 16));
}

// Pre-swizzle W^T (optionally row-concat of two matrices) into B-fragment layout with
// the column permutation matching the padded virtual layouts.
// mode 0: identity (K=I). mode 1: edge 288-layout. mode 2: node 224-layout.
__global__ void pack_b(const float* __restrict__ W0, const float* __restrict__ W1,
                       int O0, int Ot, int I, int KT, int NT, int mode, f16* __restrict__ dst){
  int t = blockIdx.x*blockDim.x + threadIdx.x;
  int total = KT*NT*32;
  if (t >= total) return;
  int lane = t & 31;
  int f    = t >> 5;
  int nt = f % NT, kt = f / NT;
  int n = nt*16 + (lane & 15);
  f16* out = dst + (size_t)f*512 + (size_t)lane*16;
#pragma unroll
  for (int hi = 0; hi < 16; ++hi){
    int k = kt*32 + ((hi>>3)<<4) + ((lane>>4)&1)*8 + (hi&7);
    int orig;
    if (mode == 1)      orig = (k<14) ? k : (k<16) ? -1 : (k<94) ? k-2 : (k<96) ? -1 : (k<288) ? k-4 : -1;
    else if (mode == 2) orig = (k<14) ? k : (k<16) ? -1 : (k<208) ? k-2 : -1;
    else                orig = (k<I) ? k : -1;
    float v = 0.f;
    if (orig >= 0 && orig < I && n < Ot)
      v = (n < O0) ? W0[(size_t)n*I + orig] : W1[(size_t)(n-O0)*I + orig];
    out[hi] = (f16)v;
  }
}

__global__ __launch_bounds__(128)
void gatedgn_kernel(Params P, const f16* __restrict__ wsp, float* __restrict__ out){
  __shared__ __align__(16) unsigned char smem[SMEM_BYTES];
  f16*   sX   = (f16*)(smem + OFF_X);
  f16*   sG16 = (f16*)(smem + OFF_G16);
  f16*   sE   = (f16*)(smem + OFF_E);
  float* sH   = (float*)(smem + OFF_H);
  float* sHs  = (float*)(smem + OFF_HS);
  float* sEs  = (float*)(smem + OFF_ES);
  float* sGv  = (float*)(smem + OFF_GV);
  f16*   sRE  = (f16*)(smem + OFF_RE);
  f16*   sUpd = (f16*)(smem + OFF_UPD);
  f16*   sT1  = (f16*)(smem + OFF_T1);
  float* sNS  = (float*)(smem + OFF_NS);
  f16*   sEdg = (f16*)(smem + OFF_EDG);
  f16*   sRH  = (f16*)(smem + OFF_RH);
  f16*   sUn  = (f16*)(smem + OFF_UN);
  f16*   sNT1 = (f16*)(smem + OFF_NT1);
  f16*   sNT2 = (f16*)(smem + OFF_NT2);
  float* sUg  = (float*)(smem + OFF_UG);
  float* sTg  = (float*)(smem + OFF_TG);
  float* sT2g = (float*)(smem + OFF_T2G);

  const int tid  = threadIdx.x;
  const int wave = tid >> 5;
  const int lane = tid & 31;
  const int gph  = blockIdx.x;
  const int col16   = lane & 15;
  const int rowHalf = (lane >> 4) << 3;

#define PF(i) ((const float*)P.p[i])
  const float* towers = PF(0) + (size_t)gph * (KN*DI);

  // ---------------- init: h = towers@E_n^T+b ; e = b_Ee ; g = b_Eg ----------------
  for (int idx = tid; idx < KN*DI; idx += 128){
    int i = idx / DI, d = idx - (idx/DI)*DI;
    sX[i*80 + d] = (f16)towers[idx];
  }
  for (int idx = tid; idx < KN*64; idx += 128){
    int i = idx >> 6, c = idx & 63;
    float acc = PF(2)[c];
    const float* W = PF(1);
#pragma unroll
    for (int d = 0; d < DI; ++d) acc += W[c*DI + d] * towers[i*DI + d];
    sH[i*64 + c] = acc;
  }
  for (int idx = tid; idx < NE*64; idx += 128) sE[idx] = (f16)PF(4)[idx & 63];
  if (tid < 64){ float v = PF(6)[tid]; sGv[tid] = v; sG16[tid] = (f16)v; }
  __syncthreads();

  const int* kp = (const int*)P.p[43];
  int ksteps = kp ? kp[0] : 3;

  for (int step = 0; step < ksteps; ++step){
    // refresh f16 h-copy inside sX (offset 16 in row)
    for (int idx = tid; idx < KN*64; idx += 128)
      sX[(idx>>6)*80 + 16 + (idx&63)] = (f16)sH[idx];
    __syncthreads();

    // ===== edge pass 1: XX(112x288) @ [M_reset;M_update]^T -> resets*e, update =====
    for (int t = wave; t < 56; t += 4){
      int mt = t >> 3, nt = t & 7;
      int r = mt*16 + col16;
      int i = r / KN, j = r - i*KN;
      v8f acc = {0.f,0.f,0.f,0.f,0.f,0.f,0.f,0.f};
#pragma unroll
      for (int kt = 0; kt < 9; ++kt){
        v16h a = gather_edge_A(sX, sG16, sE, i, j, r, kt, lane);
        v16h b = load_bfrag(wsp + WMRU, kt*8 + nt, lane);
        acc = __builtin_amdgcn_wmma_f32_16x16x32_f16(false, a, false, b, (short)0, acc, false, false);
      }
      int rowBase = mt*16 + rowHalf;
      if (nt < 4){
        int col = nt*16 + col16;
        float b = PF(16)[col];                       // M_reset bias (plain linear)
#pragma unroll
        for (int rr = 0; rr < 8; ++rr){
          int row = rowBase + rr;
          float rv = acc[rr] + b;
          sRE[row*64 + col] = (f16)(rv * (float)sE[row*64 + col]);
        }
      } else {
        int col = (nt-4)*16 + col16;
        float b = PF(14)[col];                       // M_update bias
#pragma unroll
        for (int rr = 0; rr < 8; ++rr){
          int row = rowBase + rr;
          sUpd[row*64 + col] = (f16)sigm(acc[rr] + b);
        }
      }
    }
    __syncthreads();

    // ===== edge pass 2: [base | resets*e] @ M[0]^T, relu =====
    for (int t = wave; t < 28; t += 4){
      int mt = t >> 2, nt = t & 3;
      int r = mt*16 + col16;
      int i = r / KN, j = r - i*KN;
      v8f acc = {0.f,0.f,0.f,0.f,0.f,0.f,0.f,0.f};
#pragma unroll
      for (int kt = 0; kt < 9; ++kt){
        v16h a = gather_edge_A(sX, sG16, sRE, i, j, r, kt, lane);
        v16h b = load_bfrag(wsp + WM0, kt*4 + nt, lane);
        acc = __builtin_amdgcn_wmma_f32_16x16x32_f16(false, a, false, b, (short)0, acc, false, false);
      }
      int rowBase = mt*16 + rowHalf, col = nt*16 + col16;
      float b = PF(8)[col];
#pragma unroll
      for (int rr = 0; rr < 8; ++rr)
        sT1[(rowBase+rr)*64 + col] = (f16)fmaxf(acc[rr] + b, 0.f);
    }
    __syncthreads();

    // ===== M[1], relu =====
    for (int t = wave; t < 28; t += 4){
      int mt = t >> 2, nt = t & 3;
      int r = mt*16 + col16;
      v8f acc = {0.f,0.f,0.f,0.f,0.f,0.f,0.f,0.f};
#pragma unroll
      for (int kt = 0; kt < 2; ++kt){
        v16h a = gather_row64_A(sT1, r, kt, lane);
        v16h b = load_bfrag(wsp + WM1, kt*4 + nt, lane);
        acc = __builtin_amdgcn_wmma_f32_16x16x32_f16(false, a, false, b, (short)0, acc, false, false);
      }
      int rowBase = mt*16 + rowHalf, col = nt*16 + col16;
      float b = PF(10)[col];
#pragma unroll
      for (int rr = 0; rr < 8; ++rr)
        sRE[(rowBase+rr)*64 + col] = (f16)fmaxf(acc[rr] + b, 0.f);
    }
    __syncthreads();

    // ===== M[2] + GRU gate: e = (1-u)*e + u*new_e =====
    for (int t = wave; t < 28; t += 4){
      int mt = t >> 2, nt = t & 3;
      int r = mt*16 + col16;
      v8f acc = {0.f,0.f,0.f,0.f,0.f,0.f,0.f,0.f};
#pragma unroll
      for (int kt = 0; kt < 2; ++kt){
        v16h a = gather_row64_A(sRE, r, kt, lane);
        v16h b = load_bfrag(wsp + WM2, kt*4 + nt, lane);
        acc = __builtin_amdgcn_wmma_f32_16x16x32_f16(false, a, false, b, (short)0, acc, false, false);
      }
      int rowBase = mt*16 + rowHalf, col = nt*16 + col16;
      float b = PF(12)[col];
#pragma unroll
      for (int rr = 0; rr < 8; ++rr){
        int row = rowBase + rr;
        float ne = acc[rr] + b;
        float u  = (float)sUpd[row*64 + col];
        float eo = (float)sE[row*64 + col];
        sE[row*64 + col] = (f16)((1.f - u)*eo + u*ne);
      }
    }
    __syncthreads();

    // ===== edge sums (vectorized): per-node minus diagonal, plus totals =====
    for (int idx = tid; idx < 80; idx += 128){
      int i = idx >> 3, c0 = (idx & 7) * 8;
      float s[8] = {0.f,0.f,0.f,0.f,0.f,0.f,0.f,0.f};
#pragma unroll
      for (int j = 0; j < KN; ++j){
        v8h ev = *(const v8h*)(sE + (i*KN + j)*64 + c0);
#pragma unroll
        for (int q = 0; q < 8; ++q) s[q] += (float)ev[q];
      }
      v8h dv = *(const v8h*)(sE + (i*KN + i)*64 + c0);
      v8h eg;
#pragma unroll
      for (int q = 0; q < 8; ++q){
        sNS[i*64 + c0 + q] = s[q];
        eg[q] = (f16)(s[q] - (float)dv[q]);
      }
      *(v8h*)(sEdg + i*64 + c0) = eg;
    }
    __syncthreads();
    if (tid < 64){
      float s = 0.f;
#pragma unroll
      for (int i = 0; i < KN; ++i) s += sNS[i*64 + tid];
      sEs[tid] = s;
    }
    __syncthreads();

    // ===== node pass 1: xn @ [U_reset;U_update]^T =====
    for (int nt = wave; nt < 8; nt += 4){
      int r = col16;
      v8f acc = {0.f,0.f,0.f,0.f,0.f,0.f,0.f,0.f};
#pragma unroll
      for (int kt = 0; kt < 7; ++kt){
        v16h a = gather_node_A(sX, sEdg, sG16, sX + 16, 80, r, kt, lane);
        v16h b = load_bfrag(wsp + WURU, kt*8 + nt, lane);
        acc = __builtin_amdgcn_wmma_f32_16x16x32_f16(false, a, false, b, (short)0, acc, false, false);
      }
      if (nt < 4){
        int col = nt*16 + col16;
        float b = PF(26)[col];
#pragma unroll
        for (int rr = 0; rr < 8; ++rr){
          int row = rowHalf + rr;
          float rn = sigm(acc[rr] + b);
          sRH[row*64 + col] = (f16)(rn * (float)sX[row*80 + 16 + col]);
        }
      } else {
        int col = (nt-4)*16 + col16;
        float b = PF(24)[col];
#pragma unroll
        for (int rr = 0; rr < 8; ++rr)
          sUn[(rowHalf+rr)*64 + col] = (f16)sigm(acc[rr] + b);
      }
    }
    __syncthreads();

    // ===== node pass 2: U[0], relu =====
    for (int nt = wave; nt < 4; nt += 4){
      int r = col16;
      v8f acc = {0.f,0.f,0.f,0.f,0.f,0.f,0.f,0.f};
#pragma unroll
      for (int kt = 0; kt < 7; ++kt){
        v16h a = gather_node_A(sX, sEdg, sG16, sRH, 64, r, kt, lane);
        v16h b = load_bfrag(wsp + WU0, kt*4 + nt, lane);
        acc = __builtin_amdgcn_wmma_f32_16x16x32_f16(false, a, false, b, (short)0, acc, false, false);
      }
      int col = nt*16 + col16;
      float b = PF(18)[col];
#pragma unroll
      for (int rr = 0; rr < 8; ++rr)
        sNT1[(rowHalf+rr)*64 + col] = (f16)fmaxf(acc[rr] + b, 0.f);
    }
    __syncthreads();

    // ===== U[1], relu =====
    for (int nt = wave; nt < 4; nt += 4){
      int r = col16;
      v8f acc = {0.f,0.f,0.f,0.f,0.f,0.f,0.f,0.f};
#pragma unroll
      for (int kt = 0; kt < 2; ++kt){
        v16h a = gather_row64_A(sNT1, r, kt, lane);
        v16h b = load_bfrag(wsp + WU1, kt*4 + nt, lane);
        acc = __builtin_amdgcn_wmma_f32_16x16x32_f16(false, a, false, b, (short)0, acc, false, false);
      }
      int col = nt*16 + col16;
      float b = PF(20)[col];
#pragma unroll
      for (int rr = 0; rr < 8; ++rr)
        sNT2[(rowHalf+rr)*64 + col] = (f16)fmaxf(acc[rr] + b, 0.f);
    }
    __syncthreads();

    // ===== U[2] + GRU gate on h =====
    for (int nt = wave; nt < 4; nt += 4){
      int r = col16;
      v8f acc = {0.f,0.f,0.f,0.f,0.f,0.f,0.f,0.f};
#pragma unroll
      for (int kt = 0; kt < 2; ++kt){
        v16h a = gather_row64_A(sNT2, r, kt, lane);
        v16h b = load_bfrag(wsp + WU2, kt*4 + nt, lane);
        acc = __builtin_amdgcn_wmma_f32_16x16x32_f16(false, a, false, b, (short)0, acc, false, false);
      }
      int col = nt*16 + col16;
      float b = PF(22)[col];
#pragma unroll
      for (int rr = 0; rr < 8; ++rr){
        int row = rowHalf + rr;
        float nh = acc[rr] + b;
        float u  = (float)sUn[row*64 + col];
        float ho = sH[row*64 + col];
        sH[row*64 + col] = (1.f - u)*ho + u*nh;
      }
    }
    __syncthreads();

    // hs = sum_i h
    if (tid < 64){
      float s = 0.f;
#pragma unroll
      for (int i = 0; i < KN; ++i) s += sH[i*64 + tid];
      sHs[tid] = s;
    }
    __syncthreads();

    // ===== global update (M=1, plain VALU f32) =====
    if (tid < 64){
      float ar = PF(36)[tid], au = PF(34)[tid];
      const float* Wr = PF(35); const float* Wu = PF(33);
      for (int k = 0; k < 192; ++k){
        float x = (k < 64) ? sHs[k] : (k < 128 ? sEs[k-64] : sGv[k-128]);
        ar += Wr[tid*192 + k]*x;
        au += Wu[tid*192 + k]*x;
      }
      sUg[tid] = sigm(au);
      sTg[tid] = sigm(ar) * sGv[tid];   // rg * g
    }
    __syncthreads();
    if (tid < 64){
      float acc = PF(28)[tid];
      const float* W = PF(27);
      for (int k = 0; k < 192; ++k){
        float x = (k < 64) ? sHs[k] : (k < 128 ? sEs[k-64] : sTg[k-128]);
        acc += W[tid*192 + k]*x;
      }
      sT2g[tid] = fmaxf(acc, 0.f);
    }
    __syncthreads();
    if (tid < 64){
      float acc = PF(30)[tid];
      const float* W = PF(29);
      for (int k = 0; k < 64; ++k) acc += W[tid*64 + k]*sT2g[k];
      sTg[tid] = fmaxf(acc, 0.f);
    }
    __syncthreads();
    if (tid < 64){
      float acc = PF(32)[tid];
      const float* W = PF(31);
      for (int k = 0; k < 64; ++k) acc += W[tid*64 + k]*sTg[k];
      float u = sUg[tid];
      float gn = (1.f - u)*sGv[tid] + u*acc;
      sGv[tid]  = gn;
      sG16[tid] = (f16)gn;
    }
    __syncthreads();
  } // steps

  // ===== output head: sigmoid(O2(relu(O1(relu(O0(g)))))) =====
  if (tid < 64){
    float acc = PF(38)[tid];
    const float* W = PF(37);
    for (int k = 0; k < 64; ++k) acc += W[tid*64 + k]*sGv[k];
    sTg[tid] = fmaxf(acc, 0.f);
  }
  __syncthreads();
  if (tid < 64){
    float acc = PF(40)[tid];
    const float* W = PF(39);
    for (int k = 0; k < 64; ++k) acc += W[tid*64 + k]*sTg[k];
    sT2g[tid] = fmaxf(acc, 0.f);
  }
  __syncthreads();
  if (tid == 0){
    float acc = PF(42)[0];
    const float* W = PF(41);
    for (int k = 0; k < 64; ++k) acc += W[k]*sT2g[k];
    out[gph] = 1.f/(1.f + __expf(-acc));
  }
#undef PF
}

extern "C" void kernel_launch(void* const* d_in, const int* in_sizes, int n_in,
                              void* d_out, int out_size, void* d_ws, size_t ws_size,
                              hipStream_t stream)
{
  (void)out_size; (void)ws_size;
  Params P;
  for (int i = 0; i < 48; ++i) P.p[i] = (i < n_in) ? d_in[i] : nullptr;
  f16* ws = (f16*)d_ws;

  const float* Wmr = (const float*)d_in[15];
  const float* Wmu = (const float*)d_in[13];
  const float* Wm0 = (const float*)d_in[7];
  const float* Wm1 = (const float*)d_in[9];
  const float* Wm2 = (const float*)d_in[11];
  const float* Wur = (const float*)d_in[25];
  const float* Wuu = (const float*)d_in[23];
  const float* Wu0 = (const float*)d_in[17];
  const float* Wu1 = (const float*)d_in[19];
  const float* Wu2 = (const float*)d_in[21];

#define PACK(W0, W1, O0, Ot, I, KT, NT, MODE, off) \
  do { int lanes = (KT)*(NT)*32; \
       pack_b<<<(lanes + 127)/128, 128, 0, stream>>>(W0, W1, O0, Ot, I, KT, NT, MODE, ws + (off)); } while (0)

  PACK(Wmr, Wmu, 64, 128, 284, 9, 8, 1, WMRU);
  PACK(Wm0, Wm0, 64,  64, 284, 9, 4, 1, WM0);
  PACK(Wm1, Wm1, 64,  64,  64, 2, 4, 0, WM1);
  PACK(Wm2, Wm2, 64,  64,  64, 2, 4, 0, WM2);
  PACK(Wur, Wuu, 64, 128, 206, 7, 8, 2, WURU);
  PACK(Wu0, Wu0, 64,  64, 206, 7, 4, 2, WU0);
  PACK(Wu1, Wu1, 64,  64,  64, 2, 4, 0, WU1);
  PACK(Wu2, Wu2, 64,  64,  64, 2, 4, 0, WU2);
#undef PACK

  int nGraph = in_sizes[0] / (KN*DI);
  gatedgn_kernel<<<nGraph, 128, 0, stream>>>(P, ws, (float*)d_out);
}
